// CentroidTeacher_91087666413581
// MI455X (gfx1250) — matrix operand order repaired and checked by reading, not compiled, not run on hardware
//
#include <hip/hip_runtime.h>
#include <hip/hip_bf16.h>

typedef __attribute__((ext_vector_type(16))) _Float16 v16h;
typedef __attribute__((ext_vector_type(8)))  float    v8f;
typedef unsigned int __attribute__((ext_vector_type(4))) v4u;
typedef int __attribute__((ext_vector_type(4))) v4i;
typedef int __attribute__((ext_vector_type(8))) v8i;

#define BATCH 8
#define HH 256
#define WW 256
#define MSEG 64
#define TILE 2048               // pixels staged in LDS per block iteration
#define BLOCKS_PER_BATCH 8      // reduce kernel
#define PIX_PER_BLOCK (HH*WW/BLOCKS_PER_BATCH)   // 8192
#define TILES_PER_BLOCK (PIX_PER_BLOCK/TILE)     // 4

// ---- Tensor Data Mover availability / arity probe (compile-time) ----
#if defined(__has_builtin)
# if __has_builtin(__builtin_amdgcn_tensor_load_to_lds)
#  define HAVE_TDM 1
# else
#  define HAVE_TDM 0
# endif
#else
# define HAVE_TDM 0
#endif
#if __has_include(<hip/amd_detail/amd_gfx1250_TDM.h>)
# define TDM_6ARG 1             // therock-10.0 headers -> clang-23 6-arg builtin
#else
# define TDM_6ARG 0             // ROCm 7.2 clang-22 5-arg builtin
#endif

__global__ __launch_bounds__(256) void ct_zero_ws(float* __restrict__ ws, int n) {
    int i = blockIdx.x * blockDim.x + threadIdx.x;
    if (i < n) ws[i] = 0.0f;
}

#if HAVE_TDM
// 1D contiguous TDM copy: nElems 4-byte elements, global -> LDS.
// D# per CDNA5 ISA: group0 = {flags, lds_addr, global_addr, type=2},
// group1 = {data_size=4B, tensor_dim0=nElems, tensor_dim1=1, tile_dim0=nElems, tile_dim1=1,
//           tensor_dim0_stride=nElems}. Tracked on TENSORcnt.
__device__ __forceinline__ void tdm_load_1d(void* ldsPtr, const void* gptr, unsigned nElems) {
    const unsigned long long ga = (unsigned long long)(uintptr_t)gptr;
    const unsigned lds_off = (unsigned)(uintptr_t)ldsPtr;   // LDS aperture: addr[31:0] = LDS offset
    v4u g0;
    g0.x = 1u;                                              // count=1, user descriptor
    g0.y = lds_off;                                         // lds_addr  (bits 63:32)
    g0.z = (unsigned)(ga & 0xFFFFFFFFu);                    // global_addr[31:0]
    g0.w = (unsigned)((ga >> 32) & 0x01FFFFFFu) | 0x80000000u; // global_addr[56:32] | type=2
    v8i g1;
    g1[0] = (int)(2u << 16);                                // workgroup_mask=0, data_size=2 (4B)
    g1[1] = (int)((nElems & 0xFFFFu) << 16);                // tensor_dim0[15:0]   @bits[63:48]
    g1[2] = (int)((nElems >> 16) | (1u << 16));             // tensor_dim0[31:16], tensor_dim1=1
    g1[3] = (int)((nElems & 0xFFFFu) << 16);                // tile_dim0           @bits[127:112]
    g1[4] = 1;                                              // tile_dim1=1, tile_dim2=0
    g1[5] = (int)nElems;                                    // tensor_dim0_stride[31:0]
    g1[6] = 0;
    g1[7] = 0;
    const v4i z4 = {0, 0, 0, 0};
#if TDM_6ARG
    const v8i z8 = {0, 0, 0, 0, 0, 0, 0, 0};
    __builtin_amdgcn_tensor_load_to_lds(g0, g1, z4, z4, z8, 0);
#else
    __builtin_amdgcn_tensor_load_to_lds(g0, g1, z4, z4, 0);
#endif
}
#endif

// Segmented reduction as one-hot matmul on WMMA.
// stats[b][s][0..3] = { count, motion_sum, ysum, xsum } over pixels with seg==s.
__global__ __launch_bounds__(256) void ct_reduce_wmma(const int* __restrict__ seg,
                                                      const float* __restrict__ mot,
                                                      float* __restrict__ stats) {
    __shared__ int   s_segi[TILE];   // raw int32 segment ids (TDM delivers raw)
    __shared__ float s_motf[TILE];   // raw f32 motion

    const int tid  = threadIdx.x;
    const int lane = tid & 31;
    const int wave = tid >> 5;
    const int b    = blockIdx.x >> 3;          // BLOCKS_PER_BATCH == 8
    const int part = blockIdx.x & 7;
    const int imgBase = b * (HH * WW);
    const int pixBase = imgBase + part * PIX_PER_BLOCK;

    v8f c0 = {}, c1 = {}, c2 = {}, c3 = {};    // 4 segment groups of 16

    const int m       = lane & 15;             // A-matrix row this lane supplies
    const int kbase   = (lane < 16) ? 0 : 8;   // A-matrix K half
    const int halfOff = (lane < 16) ? 0 : 8;   // B-matrix K half
    const _Float16 one_h  = (_Float16)1.0f;
    const _Float16 zero_h = (_Float16)0.0f;

    for (int t = 0; t < TILES_PER_BLOCK; ++t) {
        const int tileBase = pixBase + t * TILE;

        if (t + 1 < TILES_PER_BLOCK) {         // global_prefetch_b8 for next tile
            __builtin_prefetch(&seg[tileBase + TILE + tid * 8], 0, 0);
            __builtin_prefetch(&mot[tileBase + TILE + tid * 8], 0, 0);
        }

#if HAVE_TDM
        // Tensor Data Mover: one DMA per operand per tile, issued by wave 0.
        if (wave == 0) {
            tdm_load_1d(s_segi, &seg[tileBase], TILE);
            tdm_load_1d(s_motf, &mot[tileBase], TILE);
            __builtin_amdgcn_s_wait_tensorcnt(0);
        }
#else
        #pragma unroll
        for (int i = tid; i < TILE; i += 256) {
            s_segi[i] = seg[tileBase + i];
            s_motf[i] = mot[tileBase + i];
        }
#endif
        __syncthreads();

        // Each wave consumes 256 pixels (= one image row): 8 chunks of 32 (= WMMA K).
        #pragma unroll
        for (int ci = 0; ci < 8; ++ci) {
            const int p0  = wave * 256 + ci * 32;      // within tile
            const int gp0 = tileBase - imgBase + p0;   // pixel index within image
            const int hy  = gp0 >> 8;                  // constant across the chunk (W=256)
            const int wx0 = gp0 & 255;
            const _Float16 yv = (_Float16)(-1.0f + (float)hy * (2.0f / 255.0f));

            // ---- B (one-hot, 32x16) for the 4 segment groups ----
            // Lane = column n (= lane&15); needs seg ids at k = 8h..8h+7 and 16+8h..23+8h (broadcast LDS reads).
            const int* sp = &s_segi[p0 + halfOff];
            v16h b0 = {}, b1 = {}, b2 = {}, b3 = {};
            #pragma unroll
            for (int j = 0; j < 16; ++j) {
                const int s = sp[(j < 8) ? j : (j + 8)];
                const bool match = (s & 15) == m;
                const int  g     = s >> 4;
                b0[j] = (match && g == 0) ? one_h : zero_h;
                b1[j] = (match && g == 1) ? one_h : zero_h;
                b2[j] = (match && g == 2) ? one_h : zero_h;
                b3[j] = (match && g == 3) ? one_h : zero_h;
            }

            // ---- A (values, 16x32): row0=1, row1=motion, row2=y, row3=x ----
            v16h a = {};
            #pragma unroll
            for (int v = 0; v < 8; ++v) {
                #pragma unroll
                for (int h2 = 0; h2 < 2; ++h2) {
                    const int K = kbase + ((v < 4) ? (2 * v + h2) : (16 + 2 * (v - 4) + h2));
                    const int j = 2 * v + h2;
                    const _Float16 mv = (_Float16)s_motf[p0 + K];
                    const _Float16 xv = (_Float16)(-1.0f + (float)(wx0 + K) * (2.0f / 255.0f));
                    _Float16 val = zero_h;
                    val = (m == 0) ? one_h : val;
                    val = (m == 1) ? mv    : val;
                    val = (m == 2) ? yv    : val;
                    val = (m == 3) ? xv    : val;
                    a[j] = val;
                }
            }

            // D = A x B + C, f32 accumulate (no EXEC divergence in this loop)
            c0 = __builtin_amdgcn_wmma_f32_16x16x32_f16(false, a, false, b0, (short)0, c0, false, false);
            c1 = __builtin_amdgcn_wmma_f32_16x16x32_f16(false, a, false, b1, (short)0, c1, false, false);
            c2 = __builtin_amdgcn_wmma_f32_16x16x32_f16(false, a, false, b2, (short)0, c2, false, false);
            c3 = __builtin_amdgcn_wmma_f32_16x16x32_f16(false, a, false, b3, (short)0, c3, false, false);
        }
        __syncthreads();
    }

    // D layout: lanes 0..15 hold column (segment) = 16g + lane; VGPR r = stat row r (r=0..3).
    if (lane < 16) {
        float* sb = stats + b * MSEG * 4;
        #pragma unroll
        for (int j = 0; j < 4; ++j) {
            atomicAdd(&sb[(0 * 16 + lane) * 4 + j], c0[j]);
            atomicAdd(&sb[(1 * 16 + lane) * 4 + j], c1[j]);
            atomicAdd(&sb[(2 * 16 + lane) * 4 + j], c2[j]);
            atomicAdd(&sb[(3 * 16 + lane) * 4 + j], c3[j]);
        }
    }
}

// Finalize per-segment stats (in LDS) and scatter per-pixel outputs.
__global__ __launch_bounds__(256) void ct_output(const int* __restrict__ seg,
                                                 const float* __restrict__ stats,
                                                 float* __restrict__ out) {
    __shared__ float s_cy[MSEG];
    __shared__ float s_cx[MSEG];
    __shared__ float s_th[MSEG];

    const int tid = threadIdx.x;
    const int pix = blockIdx.x * 256 + tid;          // global pixel over B*H*W
    const int b   = blockIdx.x / (HH * WW / 256);    // 256 blocks per batch

    if (tid < MSEG) {
        const float* st = stats + (b * MSEG + tid) * 4;
        const float cnt = st[0], ms = st[1], ys = st[2], xs = st[3];
        const float denom  = fmaxf(cnt, 1.0f);
        const bool  moving = (ms / denom) > 0.5f;    // THRESH
        s_th[tid] = moving ? 1.0f : 0.0f;
        s_cy[tid] = moving ? (ys / denom) : 0.0f;    // count >= 1 whenever moving
        s_cx[tid] = moving ? (xs / denom) : 0.0f;
    }
    __syncthreads();

    const int p  = pix - b * (HH * WW);
    const int hy = p >> 8;
    const int wx = p & 255;
    const int s  = seg[pix];

    const float th = s_th[s];
    const float y  = -1.0f + (float)hy * (2.0f / 255.0f);
    const float x  = -1.0f + (float)wx * (2.0f / 255.0f);
    const float offy = (s_cy[s] - y) * 127.5f * th;  // scale = (H-1)/2
    const float offx = (s_cx[s] - x) * 127.5f * th;  // scale = (W-1)/2

    out[((b * 2 + 0) * HH + hy) * WW + wx] = offy;
    out[((b * 2 + 1) * HH + hy) * WW + wx] = offx;
    out[BATCH * 2 * HH * WW + pix]         = th;
}

extern "C" void kernel_launch(void* const* d_in, const int* in_sizes, int n_in,
                              void* d_out, int out_size, void* d_ws, size_t ws_size,
                              hipStream_t stream) {
    (void)in_sizes; (void)n_in; (void)out_size; (void)ws_size;
    const int*   seg = (const int*)d_in[0];      // (8,1,256,256) int32 in [0,64)
    const float* mot = (const float*)d_in[1];    // (8,1,1,256,256) f32
    float* out   = (float*)d_out;                // offsets (8,2,256,256) ++ thingness (8,1,256,256)
    float* stats = (float*)d_ws;                 // 8*64*4 floats = 8 KB

    ct_zero_ws<<<8, 256, 0, stream>>>(stats, BATCH * MSEG * 4);
    ct_reduce_wmma<<<BATCH * BLOCKS_PER_BATCH, 256, 0, stream>>>(seg, mot, stats);
    ct_output<<<BATCH * HH * WW / 256, 256, 0, stream>>>(seg, stats, out);
}
